// GAT_300647711304
// MI455X (gfx1250) — compile-verified, matrix operand order checked by reading, not run on hardware
//
#include <hip/hip_runtime.h>
#include <hip/hip_bf16.h>

typedef __attribute__((ext_vector_type(16))) _Float16 v16h;
typedef __attribute__((ext_vector_type(8)))  _Float16 v8h;
typedef __attribute__((ext_vector_type(8)))  float    v8f;
typedef __attribute__((ext_vector_type(4)))  unsigned int v4u;
typedef __attribute__((ext_vector_type(8)))  int      v8i;
typedef __attribute__((ext_vector_type(4)))  int      v4i;

#define ND 128
#define NEG_SLOPE 0.2f
#define BN_EPS 1e-5f

// ---------------------------------------------------------------------------
// float atomic max via int/uint sign trick (monotone bit ordering of IEEE754)
// ---------------------------------------------------------------------------
__device__ __forceinline__ void atomicMaxF(float* addr, float v) {
    if (v >= 0.0f) atomicMax((int*)addr, __float_as_int(v));
    else           atomicMin((unsigned int*)addr, __float_as_uint(v));
}

// ---------------------------------------------------------------------------
// TDM: DMA a flat 32KB (16384 x f16) block from global into LDS.
// D# packed per CDNA5 ISA 8.3/8.4: group0 = {count, lds_addr, global_addr,
// type=2}; group1 = {data_size=2B, tensor_dim0=tile_dim0=16384, stride=16384}.
// 1-D tile: tile_dim1/2 = 0 (unused). Issued by one wave; EXEC ignored by TDM.
// This toolchain uses the 6-arg builtin form (g0, g1, g2, g3, g_extra, cpol).
// ---------------------------------------------------------------------------
__device__ __forceinline__ void tdm_load_w(const _Float16* gsrc, _Float16* lds) {
    unsigned long long ga = (unsigned long long)(size_t)gsrc;
    unsigned int ldsOff   = (unsigned int)(size_t)lds;   // LDS aperture: addr[31:0]

    v4u g0;
    g0[0] = 1u;                                            // count=1 (valid user D#)
    g0[1] = ldsOff;                                        // lds_addr
    g0[2] = (unsigned int)(ga & 0xffffffffu);              // global_addr[31:0]
    g0[3] = (unsigned int)((ga >> 32) & 0x01ffffffu)       // global_addr[56:32]
          | 0x80000000u;                                   // type=2 ("image")

    v8i g1;
    g1[0] = (1 << 16);        // workgroup_mask=0, data_size=1 -> 2 bytes
    g1[1] = 0x40000000;       // tensor_dim0[15:0]=16384 -> dw1[31:16]
    g1[2] = 0x00010000;       // tensor_dim0[31:16]=0, tensor_dim1=1 -> dw2[31:16]
    g1[3] = 0x40000000;       // tensor_dim1 hi=0, tile_dim0=16384 -> dw3[31:16]
    g1[4] = 0;                // tile_dim1=0, tile_dim2=0 (1-D tile)
    g1[5] = 16384;            // tensor_dim0_stride lo
    g1[6] = 0;                // stride0 hi, stride1 lo
    g1[7] = 0;                // stride1 hi

    v4i z4; z4[0] = 0; z4[1] = 0; z4[2] = 0; z4[3] = 0;
    v8i z8;
#pragma unroll
    for (int i = 0; i < 8; ++i) z8[i] = 0;

    __builtin_amdgcn_tensor_load_to_lds(g0, g1, z4, z4, z8, 0);
}

// ---------------------------------------------------------------------------
// Kc: one-shot conversion of all L=3 weight matrices to f16, transposed into
// the exact LDS layout the GEMM wants: Wt16[l][n*128 + k] = (f16) W[l][k][n].
// ---------------------------------------------------------------------------
__global__ void gat_convW(const float* __restrict__ Ws, _Float16* __restrict__ Wt16)
{
    int idx = blockIdx.x * blockDim.x + threadIdx.x;
    if (idx >= 3 * ND * ND) return;
    int l = idx >> 14;
    int r = idx & (ND * ND - 1);
    int k = r >> 7, n = r & 127;
    Wt16[l * ND * ND + n * ND + k] = (_Float16)Ws[idx];
}

// ---------------------------------------------------------------------------
// K1: h = X @ W via v_wmma_f32_16x16x32_f16, fused alpha_src/alpha_dst dots.
// Block = 256 threads = 8 wave32; block tile = 128 rows x 128 cols.
// W tile (pre-converted f16, transposed) arrives in LDS via TDM.
// ---------------------------------------------------------------------------
__global__ __launch_bounds__(256) void gat_gemm_wmma(
    const float* __restrict__ Xin, const _Float16* __restrict__ Wt16,
    const float* __restrict__ a_s, const float* __restrict__ a_d,
    float* __restrict__ H, float* __restrict__ alphaS, float* __restrict__ alphaD,
    int Nnodes)
{
    __shared__ _Float16 Wt[ND * ND];        // Wt[n*128 + k]

    if (threadIdx.x < 32) {                 // uniform per-wave branch: wave 0 only
        tdm_load_w(Wt16, Wt);
        __builtin_amdgcn_s_wait_tensorcnt(0);
    }
    __syncthreads();

    const int tid   = threadIdx.x;
    const int lane  = tid & 31;
    const int wave  = tid >> 5;
    const int nl    = lane & 15;            // N (or M for A-frag) within tile
    const int half  = lane >> 4;            // lane-group select
    const int khalf = half * 8;

    const int rowTile = blockIdx.x * 128 + wave * 16;

    // A-fragment source row for this lane (clamp: keeps EXEC uniform for WMMA)
    int arow  = rowTile + nl;
    int arowC = arow < Nnodes ? arow : (Nnodes - 1);
    const float* xrow = Xin + (size_t)arowC * ND;

    v8f acc[8];
#pragma unroll
    for (int t = 0; t < 8; ++t) acc[t] = (v8f)0.0f;

#pragma unroll
    for (int kk = 0; kk < ND; kk += 32) {
        // ---- A fragment: lane holds row=nl, K = kk+khalf+[0..8) and kk+16+khalf+[0..8)
        const float4* p0 = (const float4*)(xrow + kk + khalf);
        const float4* p1 = (const float4*)(xrow + kk + 16 + khalf);
        float4 f0 = p0[0], f1 = p0[1], f2 = p1[0], f3 = p1[1];
        v16h afrag;
        afrag[0]  = (_Float16)f0.x; afrag[1]  = (_Float16)f0.y;
        afrag[2]  = (_Float16)f0.z; afrag[3]  = (_Float16)f0.w;
        afrag[4]  = (_Float16)f1.x; afrag[5]  = (_Float16)f1.y;
        afrag[6]  = (_Float16)f1.z; afrag[7]  = (_Float16)f1.w;
        afrag[8]  = (_Float16)f2.x; afrag[9]  = (_Float16)f2.y;
        afrag[10] = (_Float16)f2.z; afrag[11] = (_Float16)f2.w;
        afrag[12] = (_Float16)f3.x; afrag[13] = (_Float16)f3.y;
        afrag[14] = (_Float16)f3.z; afrag[15] = (_Float16)f3.w;

#pragma unroll
        for (int t = 0; t < 8; ++t) {
            // ---- B fragment: lane holds col n = 16t+nl, same K split as A
            const v8h* bp0 = (const v8h*)&Wt[(16 * t + nl) * ND + kk + khalf];
            const v8h* bp1 = (const v8h*)&Wt[(16 * t + nl) * ND + kk + 16 + khalf];
            v8h b0 = *bp0, b1 = *bp1;
            v16h bfrag;
#pragma unroll
            for (int j = 0; j < 8; ++j) { bfrag[j] = b0[j]; bfrag[8 + j] = b1[j]; }

            acc[t] = __builtin_amdgcn_wmma_f32_16x16x32_f16(
                false, afrag, false, bfrag, (short)0, acc[t], false, false);
        }
    }

    // ---- store H: C/D layout -> lane n=nl, vgpr r -> m = 8*half + r
#pragma unroll
    for (int r = 0; r < 8; ++r) {
        int orow = rowTile + half * 8 + r;
        if (orow < Nnodes) {
            float* hp = H + (size_t)orow * ND + nl;
#pragma unroll
            for (int t = 0; t < 8; ++t) hp[16 * t] = acc[t][r];
        }
    }

    // ---- fused attention dots: alpha[row] = sum_n h[row][n]*a[n]
    float asv[8], adv[8];
#pragma unroll
    for (int t = 0; t < 8; ++t) { asv[t] = a_s[16 * t + nl]; adv[t] = a_d[16 * t + nl]; }

#pragma unroll
    for (int r = 0; r < 8; ++r) {
        float vs = 0.0f, vd = 0.0f;
#pragma unroll
        for (int t = 0; t < 8; ++t) { vs += acc[t][r] * asv[t]; vd += acc[t][r] * adv[t]; }
        // reduce over the 16 lanes sharing this row (xor bits 0..3 stay in-half)
#pragma unroll
        for (int off = 1; off <= 8; off <<= 1) {
            vs += __shfl_xor(vs, off, 32);
            vd += __shfl_xor(vd, off, 32);
        }
        int orow = rowTile + half * 8 + r;
        if (nl == 0 && orow < Nnodes) { alphaS[orow] = vs; alphaD[orow] = vd; }
    }
}

// ---------------------------------------------------------------------------
// K0: per-layer init: out = bias (broadcast), mmax = -inf, ssum = 0, stats = 0
// ---------------------------------------------------------------------------
__global__ void gat_init(float* __restrict__ out, const float* __restrict__ bias,
                         float* __restrict__ mmax, float* __restrict__ ssum,
                         float* __restrict__ stats, int total, int Nnodes)
{
    int idx = blockIdx.x * blockDim.x + threadIdx.x;
    if (idx < total)   out[idx] = bias[idx & (ND - 1)];
    if (idx < Nnodes) { mmax[idx] = -__builtin_huge_valf(); ssum[idx] = 0.0f; }
    if (idx < 2 * ND)  stats[idx] = 0.0f;
}

// ---------------------------------------------------------------------------
// K3: e = leaky_relu(aS[src] + aD[dst]); segment max over dst
// ---------------------------------------------------------------------------
__global__ void gat_edge_max(const int* __restrict__ src, const int* __restrict__ dst,
                             const float* __restrict__ aS, const float* __restrict__ aD,
                             float* __restrict__ ew, float* __restrict__ mmax, int E)
{
    int i = blockIdx.x * blockDim.x + threadIdx.x;
    if (i >= E) return;
    int s = src[i], d = dst[i];
    float e = aS[s] + aD[d];
    e = e > 0.0f ? e : NEG_SLOPE * e;
    ew[i] = e;
    atomicMaxF(&mmax[d], e);
}

// ---------------------------------------------------------------------------
// K4: w = exp(e - m[dst]) in place; segment sum over dst
// ---------------------------------------------------------------------------
__global__ void gat_edge_exp(const int* __restrict__ dst, float* __restrict__ ew,
                             const float* __restrict__ mmax, float* __restrict__ ssum, int E)
{
    int i = blockIdx.x * blockDim.x + threadIdx.x;
    if (i >= E) return;
    int d = dst[i];
    float w = __expf(ew[i] - mmax[d]);
    ew[i] = w;
    atomicAdd(&ssum[d], w);
}

// ---------------------------------------------------------------------------
// K5: out[dst] += (w / (s[dst]+eps)) * h[src]   — one wave32 per edge,
// each lane owns 4 channels (float4 gather + 4 atomic f32 adds)
// ---------------------------------------------------------------------------
__global__ __launch_bounds__(256) void gat_aggregate(
    const int* __restrict__ src, const int* __restrict__ dst,
    const float* __restrict__ H, const float* __restrict__ ew,
    const float* __restrict__ ssum, float* __restrict__ out, int E)
{
    int gw   = (blockIdx.x * 256 + threadIdx.x) >> 5;   // edge id
    int lane = threadIdx.x & 31;
    if (gw >= E) return;
    int s = src[gw], d = dst[gw];
    float a = ew[gw] / (ssum[d] + 1e-16f);
    float4 v = ((const float4*)(H + (size_t)s * ND))[lane];
    float* o = out + (size_t)d * ND + lane * 4;
    atomicAdd(o + 0, a * v.x);
    atomicAdd(o + 1, a * v.y);
    atomicAdd(o + 2, a * v.z);
    atomicAdd(o + 3, a * v.w);
}

// ---------------------------------------------------------------------------
// K6: per-channel sum / sumsq (block-strided rows, coalesced, atomic finish)
// ---------------------------------------------------------------------------
__global__ __launch_bounds__(128) void gat_bn_stats(const float* __restrict__ x,
                                                    float* __restrict__ stats, int Nnodes)
{
    int c = threadIdx.x;
    float s = 0.0f, q = 0.0f;
    for (int r = blockIdx.x; r < Nnodes; r += gridDim.x) {
        float v = x[(size_t)r * ND + c];
        s += v; q += v * v;
    }
    atomicAdd(&stats[c], s);
    atomicAdd(&stats[ND + c], q);
}

// ---------------------------------------------------------------------------
// K7: x = relu(gamma*(x-mu)*rsqrt(var+eps)+beta), in place
// ---------------------------------------------------------------------------
__global__ void gat_bn_apply(float* __restrict__ x, const float* __restrict__ stats,
                             const float* __restrict__ gamma, const float* __restrict__ beta,
                             int total, float invN)
{
    int idx = blockIdx.x * blockDim.x + threadIdx.x;
    if (idx >= total) return;
    int c = idx & (ND - 1);
    float mu  = stats[c] * invN;
    float var = stats[ND + c] * invN - mu * mu;
    float v = x[idx];
    v = gamma[c] * (v - mu) * rsqrtf(var + BN_EPS) + beta[c];
    x[idx] = v > 0.0f ? v : 0.0f;
}

// ---------------------------------------------------------------------------
extern "C" void kernel_launch(void* const* d_in, const int* in_sizes, int n_in,
                              void* d_out, int out_size, void* d_ws, size_t ws_size,
                              hipStream_t stream)
{
    const float* x      = (const float*)d_in[0];
    const int*   ei     = (const int*)  d_in[1];
    const float* Ws     = (const float*)d_in[2];
    const float* a_src  = (const float*)d_in[3];
    const float* a_dst  = (const float*)d_in[4];
    const float* biases = (const float*)d_in[5];
    const float* gammas = (const float*)d_in[6];
    const float* betas  = (const float*)d_in[7];

    const int N = in_sizes[0] / ND;
    const int E = in_sizes[1] / 2;
    const int* src = ei;
    const int* dst = ei + E;

    // ---- workspace carve-up ----
    float* wsf  = (float*)d_ws;
    float* H    = wsf;                         // N*ND floats
    float* FEAT = H    + (size_t)N * ND;       // N*ND
    float* AS   = FEAT + (size_t)N * ND;       // N
    float* AD   = AS + N;                      // N
    float* MM   = AD + N;                      // N
    float* SS   = MM + N;                      // N
    float* EW   = SS + N;                      // E
    float* ST   = EW + E;                      // 2*ND
    _Float16* WT16 = (_Float16*)(ST + 2 * ND); // 3 * ND*ND halfs (96KB)

    const int totalND  = N * ND;
    const int gInit    = (totalND + 255) / 256;
    const int gGemm    = (N + 127) / 128;
    const int gEdge    = (E + 255) / 256;
    const int gAgg     = (E * 32 + 255) / 256; // one wave32 per edge
    const float invN   = 1.0f / (float)N;

    // one-shot: all 3 weight matrices -> f16, transposed LDS layout
    gat_convW<<<(3 * ND * ND + 255) / 256, 256, 0, stream>>>(Ws, WT16);

    for (int l = 0; l < 3; ++l) {
        const float* in  = (l == 0) ? x : FEAT;
        float*       out = (l == 2) ? (float*)d_out : FEAT;   // FEAT dead after GEMM

        gat_gemm_wmma<<<gGemm, 256, 0, stream>>>(
            in, WT16 + (size_t)l * ND * ND, a_src + l * ND, a_dst + l * ND,
            H, AS, AD, N);

        gat_init<<<gInit, 256, 0, stream>>>(out, biases + l * ND, MM, SS, ST, totalND, N);

        gat_edge_max<<<gEdge, 256, 0, stream>>>(src, dst, AS, AD, EW, MM, E);
        gat_edge_exp<<<gEdge, 256, 0, stream>>>(dst, EW, MM, SS, E);
        gat_aggregate<<<gAgg, 256, 0, stream>>>(src, dst, H, EW, SS, out, E);

        if (l < 2) {
            gat_bn_stats<<<512, 128, 0, stream>>>(FEAT, ST, N);
            gat_bn_apply<<<gInit, 256, 0, stream>>>(FEAT, ST, gammas + l * ND,
                                                    betas + l * ND, totalND, invN);
        }
    }
}
